// CenterBasedLoss_36979668419191
// MI455X (gfx1250) — matrix operand order repaired and checked by reading, not compiled, not run on hardware
//
#include <hip/hip_runtime.h>

// CenterBasedLoss on MI455X (gfx1250, wave32).
//
// segment-sum = OneHot(labels)^T @ features, computed per 16-class bucket with
// V_WMMA_F32_16X16X32_F16 after a counting-sort of sample indices by label/16.
// Memory-bound (~256MB feature stream @ 23.3TB/s => ~11us floor). Staging uses
// GLOBAL_LOAD_ASYNC_TO_LDS_B128 (ASYNCcnt) — the CDNA5 async global->LDS path.
//
#define NUM_CLASSES 1000
#define FEAT_DIM    256
#define ALPHA       0.5f
#define TILE_C      16            // classes per WMMA tile
#define NUM_TILES   63            // ceil(1000/16); 63*16 = 1008 padded classes
#define PAD_C       (NUM_TILES * TILE_C)
#define GROUP       32            // samples per WMMA K-step (K=32 for f16 WMMA)
#define SLICES      32            // parallel slices per bucket (63*32 = 2016 WGs)

typedef __attribute__((ext_vector_type(16))) _Float16 v16h;
typedef __attribute__((ext_vector_type(8)))  float    v8f;

// ---------------- workspace layout (bytes) ----------------
// [0,      4096)  : counts[1024]            (int)
// [4096,   4352)  : cursors[64]             (int)
// [4352,   4608)  : base[64]                (int, exclusive prefix; base[63]=N)
// [4608,   +1008*256*4) : sums[1008*256]    (float)
// then            : order[N]                (int)
#define WS_COUNTS  0
#define WS_CURSORS 4096
#define WS_BASE    4352
#define WS_SUMS    4608
#define WS_ORDER   (4608 + PAD_C * FEAT_DIM * 4)

// ---- CDNA5 async global->LDS copy helpers ------------------------------
// GVS addressing: mem = SGPR64(base) + VGPR32(byte_off); per-lane 16B into
// LDS[lds_byte_addr]. Tracked by ASYNCcnt.
static __device__ __forceinline__ void async_copy_b128(unsigned lds_addr,
                                                       const float* sbase,
                                                       int byte_off) {
    asm volatile("global_load_async_to_lds_b128 %0, %1, %2"
                 :: "v"(lds_addr), "v"(byte_off), "s"(sbase)
                 : "memory");
}
static __device__ __forceinline__ void async_wait0() {
#if __has_builtin(__builtin_amdgcn_s_wait_asynccnt)
    __builtin_amdgcn_s_wait_asynccnt(0);
#else
    asm volatile("s_wait_asynccnt 0x0" ::: "memory");
#endif
}

__global__ __launch_bounds__(256) void cbl_zero(int* p, int n) {
    int i = blockIdx.x * 256 + threadIdx.x;
    if (i < n) p[i] = 0;
}

__global__ __launch_bounds__(256) void cbl_count(const int* __restrict__ labels,
                                                 int* __restrict__ counts, int n) {
    int i = blockIdx.x * 256 + threadIdx.x;
    if (i < n) atomicAdd(&counts[labels[i]], 1);
}

// Tiny serial prefix over 63 buckets; negligible cost.
__global__ void cbl_prefix(const int* __restrict__ counts,
                           int* __restrict__ base, int* __restrict__ cursors) {
    if (threadIdx.x == 0) {
        int run = 0;
        for (int t = 0; t < NUM_TILES; ++t) {
            base[t]    = run;
            cursors[t] = run;
            int s = 0;
            for (int k = 0; k < TILE_C; ++k) s += counts[t * TILE_C + k];
            run += s;
        }
        base[NUM_TILES] = run;   // == N
    }
}

__global__ __launch_bounds__(256) void cbl_scatter(const int* __restrict__ labels,
                                                   int* __restrict__ cursors,
                                                   int* __restrict__ order, int n) {
    int i = blockIdx.x * 256 + threadIdx.x;
    if (i < n) {
        int t = labels[i] >> 4;                 // label / 16
        int pos = atomicAdd(&cursors[t], 1);
        order[pos] = i;
    }
}

// One WG = (bucket t, slice s). 8 waves; wave w owns feature d-tiles 2w, 2w+1.
// Per 32-sample group: async-stage rows to LDS, build one-hot A (f16),
// two V_WMMA_F32_16X16X32_F16 per wave accumulating class-sums in f32.
__global__ __launch_bounds__(256) void cbl_wmma_sums(const float* __restrict__ feat,
                                                     const int* __restrict__ labels,
                                                     const int* __restrict__ order,
                                                     const int* __restrict__ base,
                                                     float* __restrict__ sums) {
    __shared__ float ldsF[GROUP][FEAT_DIM];   // 32 KB staged feature rows
    __shared__ int   ldsL[GROUP];             // staged labels (-1 = padding)
    __shared__ int   ldsI[GROUP];             // staged sample indices

    const int t     = blockIdx.y;
    const int slice = blockIdx.x;
    const int b0 = base[t], b1 = base[t + 1];
    const int len   = b1 - b0;
    const int chunk = (len + SLICES - 1) / SLICES;
    const int start = b0 + slice * chunk;
    int end = start + chunk; if (end > b1) end = b1;

    const int  tid  = threadIdx.x;
    const int  lane = tid & 31;
    const int  wave = tid >> 5;
    const int  c0   = t * TILE_C;
    const int  mrow = lane & 15;               // M (class) / N (dim) sub-index
    const bool hi   = lane >= 16;
    const int  d0a  = (wave * 2)     * 16;
    const int  d0b  = (wave * 2 + 1) * 16;

    v8f acc0 = {}; v8f acc1 = {};

    for (int g = start; g < end; g += GROUP) {
        int cnt = end - g; if (cnt > GROUP) cnt = GROUP;

        __syncthreads();                       // prior-iteration LDS reads done
        if (tid < GROUP) {
            int idx = (tid < cnt) ? order[g + tid] : -1;
            ldsI[tid] = idx;
            ldsL[tid] = (idx >= 0) ? labels[idx] : -1;
        }
        __syncthreads();

        // Async-stage 32 rows x 1KB: 8 passes x (4 rows x 64 lanes x 16B),
        // fully coalesced, straight into LDS (no VGPR round-trip). Padding
        // rows reload row 0 (finite data); their one-hot column is 0 so they
        // contribute exactly 0 to the WMMA accumulate.
        for (int p = 0; p < 8; ++p) {
            int r    = p * 4 + (tid >> 6);
            int col4 = (tid & 63);            // float4 index within row
            int idx  = ldsI[r]; if (idx < 0) idx = 0;
            unsigned lds_addr = (unsigned)(uintptr_t)&ldsF[r][col4 * 4];
            int goff = idx * (FEAT_DIM * 4) + col4 * 16;
            async_copy_b128(lds_addr, feat, goff);
        }
        async_wait0();
        __syncthreads();

        // A: 16x32 one-hot, f16. A layout: lane<16 -> K = 0..15,
        // lane>=16 -> K = 16..31; M = lane&15.
        v16h a;
#pragma unroll
        for (int e = 0; e < 16; ++e) {
            int ka = hi ? e + 16 : e;
            a[e] = (_Float16)((ldsL[ka] == c0 + mrow) ? 1.0f : 0.0f);
        }
        // B: 32x16, f16. Lane<16 holds K=0..15, lane>=16 holds K=16..31; N = lane&15.
        v16h bA, bB;
#pragma unroll
        for (int e = 0; e < 16; ++e) {
            int kb = hi ? e + 16 : e;
            bA[e] = (_Float16)ldsF[kb][d0a + mrow];
            bB[e] = (_Float16)ldsF[kb][d0b + mrow];
        }
        acc0 = __builtin_amdgcn_wmma_f32_16x16x32_f16(false, a, false, bA,
                                                      (short)0, acc0, false, false);
        acc1 = __builtin_amdgcn_wmma_f32_16x16x32_f16(false, a, false, bB,
                                                      (short)0, acc1, false, false);
    }

    // Merge partial sums. C/D layout: VGPR v -> M = v (lanes 0-15) / v+8 (16-31),
    // N = lane&15. 63*32*4096 = 8.3M atomics total (vs 67M naive scatter).
#pragma unroll
    for (int v = 0; v < 8; ++v) {
        int c = c0 + v + (hi ? 8 : 0);
        atomicAdd(&sums[c * FEAT_DIM + d0a + mrow], acc0[v]);
        atomicAdd(&sums[c * FEAT_DIM + d0b + mrow], acc1[v]);
    }
}

__global__ __launch_bounds__(256) void cbl_finalize(const float* __restrict__ centers,
                                                    const int* __restrict__ counts,
                                                    const float* __restrict__ sums,
                                                    float* __restrict__ out, int total) {
    int i = blockIdx.x * 256 + threadIdx.x;
    if (i >= total) return;
    int c = i >> 8;                     // FEAT_DIM == 256
    float ctr = centers[i];
    int cnt = counts[c];
    if (cnt > 0) {
        float mean = sums[i] / (float)cnt;
        out[i] = (1.0f - ALPHA) * ctr + ALPHA * mean;
    } else {
        out[i] = ctr;
    }
}

extern "C" void kernel_launch(void* const* d_in, const int* in_sizes, int n_in,
                              void* d_out, int out_size, void* d_ws, size_t ws_size,
                              hipStream_t stream) {
    const float* feat    = (const float*)d_in[0];
    const int*   labels  = (const int*)d_in[1];
    const float* centers = (const float*)d_in[2];
    float*       out     = (float*)d_out;
    const int n = in_sizes[1];          // N = number of samples

    char*  ws      = (char*)d_ws;       // needs ~2.1 MB
    int*   counts  = (int*)(ws + WS_COUNTS);
    int*   cursors = (int*)(ws + WS_CURSORS);
    int*   base    = (int*)(ws + WS_BASE);
    float* sums    = (float*)(ws + WS_SUMS);
    int*   order   = (int*)(ws + WS_ORDER);

    const int zeroWords = WS_ORDER / 4;           // counts..sums inclusive
    cbl_zero   <<<(zeroWords + 255) / 256, 256, 0, stream>>>((int*)ws, zeroWords);
    cbl_count  <<<(n + 255) / 256,         256, 0, stream>>>(labels, counts, n);
    cbl_prefix <<<1, 32, 0, stream>>>(counts, base, cursors);
    cbl_scatter<<<(n + 255) / 256,         256, 0, stream>>>(labels, cursors, order, n);
    cbl_wmma_sums<<<dim3(SLICES, NUM_TILES), 256, 0, stream>>>(feat, labels, order, base, sums);
    const int total = NUM_CLASSES * FEAT_DIM;     // 256000 == out_size
    cbl_finalize<<<(total + 255) / 256,    256, 0, stream>>>(centers, counts, sums, out, total);
}
